// Net_12378095747615
// MI455X (gfx1250) — compile-verified, matrix operand order checked by reading, not run on hardware
//
#include <hip/hip_runtime.h>
#include <hip/hip_bf16.h>

#define N_NODES    100000
#define N_EDGES    1600000
#define HID        128
#define NC         3
#define NUM_GRAPHS 256

typedef __bf16 bf16x16 __attribute__((ext_vector_type(16)));
typedef float  f32x8   __attribute__((ext_vector_type(8)));

// ---------------------------------------------------------------- utilities
__global__ void zero_f32(float* __restrict__ p, int n) {
    int i = blockIdx.x * 256 + threadIdx.x;
    if (i < n) p[i] = 0.0f;
}

// Convert NC x HID x HID f32 weights -> bf16, transposed to [i][n][k] so that
// a WMMA B-fragment column is a contiguous run of K values.
__global__ void conv_wT(const float* __restrict__ W, __bf16* __restrict__ WT) {
    int t = blockIdx.x * 256 + threadIdx.x;
    if (t >= NC * HID * HID) return;
    int i = t / (HID * HID);
    int r = t % (HID * HID);
    int k = r / HID;
    int n = r % HID;
    WT[(size_t)i * HID * HID + (size_t)n * HID + k] = (__bf16)W[t];
}

// ---------------------------------------------------------------- scatter-add
// agg[dst[e]] += x[src[e]]  ; 32 lanes per edge, float4 per lane (128 ch)
__global__ void scatter_add(const float* __restrict__ x,
                            const int*  __restrict__ src,
                            const int*  __restrict__ dst,
                            float* __restrict__ agg) {
    int e = blockIdx.x * 8 + (threadIdx.x >> 5);
    if (e >= N_EDGES) return;
    int lane = threadIdx.x & 31;
    int s = src[e], d = dst[e];
    const float4 v = *(const float4*)(x + (size_t)s * HID + lane * 4);
    float* o = agg + (size_t)d * HID + lane * 4;
    atomicAdd(o + 0, v.x);
    atomicAdd(o + 1, v.y);
    atomicAdd(o + 2, v.z);
    atomicAdd(o + 3, v.w);
}

// ---------------------------------------------------------------- GraphConv layer
// out = tanh(agg @ W_rel + b_rel + x @ W_root)
// Block: 256 threads = 8 waves. Block -> 16 output rows, wave w -> cols [16w,16w+16).
// K = 128 consumed as 4 chunks of 32 via v_wmma_f32_16x16x32_bf16 (two chained
// WMMAs per chunk: one per input matrix, shared f32 accumulator).
__launch_bounds__(256, 2)
__global__ void gc_layer(const float*  __restrict__ agg,
                         const float*  __restrict__ xin,
                         const __bf16* __restrict__ WrelT,   // [n][k] bf16
                         const __bf16* __restrict__ WrootT,  // [n][k] bf16
                         const float*  __restrict__ brel,    // [HID]
                         float* __restrict__ out) {
    __shared__ __bf16 As[16][HID + 8];   // +8 bf16 pad -> 16B row skew
    __shared__ __bf16 Xs[16][HID + 8];

    const int rowbase = blockIdx.x * 16;

    // Cooperative load + f32->bf16 convert: 2048 elems each, 8 per thread.
    {
        int idx = threadIdx.x * 8;
        int row = idx >> 7;
        int col = idx & 127;
        const float* ag = agg + (size_t)(rowbase + row) * HID + col;
        const float* xg = xin + (size_t)(rowbase + row) * HID + col;
#pragma unroll
        for (int j = 0; j < 8; ++j) {
            As[row][col + j] = (__bf16)ag[j];
            Xs[row][col + j] = (__bf16)xg[j];
        }
    }
    __syncthreads();

    const int wave  = threadIdx.x >> 5;
    const int lane  = threadIdx.x & 31;
    const int ncol  = wave * 16 + (lane & 15);     // B/C/D: N = lane (fixed per lane)
    const int arow  = lane & 15;                   // A: M = lane (mod 16)
    const int khalf = (lane & 16) ? 8 : 0;         // ISA 16-bit fragment K-half split

    const __bf16* wr = WrelT  + (size_t)ncol * HID;
    const __bf16* wo = WrootT + (size_t)ncol * HID;

    f32x8 c = {};
#pragma unroll
    for (int kc = 0; kc < HID; kc += 32) {
        const int kb = kc + khalf;
        bf16x16 a, xv, br, bo;
#pragma unroll
        for (int j = 0; j < 8; ++j) {
            // VGPR0-3: K = kb..kb+7 ; VGPR4-7: K = kb+16..kb+23 (ISA 7.12.2)
            a[j]      = As[arow][kb + j];
            a[8 + j]  = As[arow][kb + 16 + j];
            xv[j]     = Xs[arow][kb + j];
            xv[8 + j] = Xs[arow][kb + 16 + j];
            br[j]     = wr[kb + j];
            br[8 + j] = wr[kb + 16 + j];
            bo[j]     = wo[kb + j];
            bo[8 + j] = wo[kb + 16 + j];
        }
        c = __builtin_amdgcn_wmma_f32_16x16x32_bf16(false, a,  false, br, (short)0, c, false, false);
        c = __builtin_amdgcn_wmma_f32_16x16x32_bf16(false, xv, false, bo, (short)0, c, false, false);
    }

    const float bias = brel[ncol];
    const int   moff = (lane & 16) ? 8 : 0;        // C/D: VGPR j -> M = j (+8 hi lanes)
#pragma unroll
    for (int j = 0; j < 8; ++j) {
        out[(size_t)(rowbase + j + moff) * HID + ncol] = tanhf(c[j] + bias);
    }
}

// ---------------------------------------------------------------- pooling
__global__ void pool_add(const float* __restrict__ x,
                         const int*  __restrict__ batch,
                         float* __restrict__ g) {
    int node = blockIdx.x * 8 + (threadIdx.x >> 5);
    if (node >= N_NODES) return;
    int lane = threadIdx.x & 31;
    int b = batch[node];
    const float4 v = *(const float4*)(x + (size_t)node * HID + lane * 4);
    float* o = g + (size_t)b * HID + lane * 4;
    atomicAdd(o + 0, v.x);
    atomicAdd(o + 1, v.y);
    atomicAdd(o + 2, v.z);
    atomicAdd(o + 3, v.w);
}

// ---------------------------------------------------------------- tiny MLP head
__global__ void mlp1(const float* __restrict__ g, const float* __restrict__ w1,
                     const float* __restrict__ b1, float* __restrict__ h) {
    int gr = blockIdx.x, n = threadIdx.x;   // grid NUM_GRAPHS, block HID
    const float* row = g + (size_t)gr * HID;
    float acc = b1[n];
    for (int k = 0; k < HID; ++k) acc = fmaf(row[k], w1[k * HID + n], acc);
    h[(size_t)gr * HID + n] = fmaxf(acc, 0.0f);
}

__global__ void mlp2(const float* __restrict__ h, const float* __restrict__ w2,
                     const float* __restrict__ b2, float* __restrict__ out) {
    int gr = threadIdx.x;                   // 1 block of NUM_GRAPHS threads
    const float* row = h + (size_t)gr * HID;
    float acc = b2[0];
    for (int k = 0; k < HID; ++k) acc = fmaf(row[k], w2[k], acc);
    out[gr] = 1.0f / (1.0f + expf(-acc));
}

// ---------------------------------------------------------------- launch
extern "C" void kernel_launch(void* const* d_in, const int* in_sizes, int n_in,
                              void* d_out, int out_size, void* d_ws, size_t ws_size,
                              hipStream_t stream) {
    const float* x      = (const float*)d_in[0];
    const float* W_rel  = (const float*)d_in[1];
    const float* b_rel  = (const float*)d_in[2];
    const float* W_root = (const float*)d_in[3];
    const float* mw1    = (const float*)d_in[4];
    const float* mb1    = (const float*)d_in[5];
    const float* mw2    = (const float*)d_in[6];
    const float* mb2    = (const float*)d_in[7];
    const int*   edges  = (const int*)d_in[8];   // [2, N_EDGES] flat
    const int*   batch  = (const int*)d_in[9];
    float*       out    = (float*)d_out;

    // workspace carve-up (256B aligned)
    char*  ws  = (char*)d_ws;
    size_t off = 0;
    auto carve = [&](size_t bytes) {
        size_t o = off;
        off = (off + bytes + 255) & ~(size_t)255;
        return o;
    };
    const size_t featB = (size_t)N_NODES * HID * sizeof(float);
    float*  agg    = (float*)(ws + carve(featB));
    float*  buf0   = (float*)(ws + carve(featB));
    float*  buf1   = (float*)(ws + carve(featB));
    __bf16* wrelT  = (__bf16*)(ws + carve((size_t)NC * HID * HID * sizeof(__bf16)));
    __bf16* wrootT = (__bf16*)(ws + carve((size_t)NC * HID * HID * sizeof(__bf16)));
    float*  gpool  = (float*)(ws + carve((size_t)NUM_GRAPHS * HID * sizeof(float)));
    float*  hbuf   = (float*)(ws + carve((size_t)NUM_GRAPHS * HID * sizeof(float)));
    (void)ws_size; (void)in_sizes; (void)n_in; (void)out_size;

    const int wElems = NC * HID * HID;
    conv_wT<<<(wElems + 255) / 256, 256, 0, stream>>>(W_rel,  wrelT);
    conv_wT<<<(wElems + 255) / 256, 256, 0, stream>>>(W_root, wrootT);

    const float* cur = x;
    float* pong[2] = {buf0, buf1};
    for (int i = 0; i < NC; ++i) {
        zero_f32<<<(N_NODES * HID + 255) / 256, 256, 0, stream>>>(agg, N_NODES * HID);
        scatter_add<<<(N_EDGES + 7) / 8, 256, 0, stream>>>(cur, edges, edges + N_EDGES, agg);
        float* nxt = pong[i & 1];
        gc_layer<<<N_NODES / 16, 256, 0, stream>>>(
            agg, cur,
            wrelT  + (size_t)i * HID * HID,
            wrootT + (size_t)i * HID * HID,
            b_rel + (size_t)i * HID,
            nxt);
        cur = nxt;
    }

    zero_f32<<<(NUM_GRAPHS * HID + 255) / 256, 256, 0, stream>>>(gpool, NUM_GRAPHS * HID);
    pool_add<<<(N_NODES + 7) / 8, 256, 0, stream>>>(cur, batch, gpool);
    mlp1<<<NUM_GRAPHS, HID, 0, stream>>>(gpool, mw1, mb1, hbuf);
    mlp2<<<1, NUM_GRAPHS, 0, stream>>>(hbuf, mw2, mb2, out);
}
